// AttentionBlock_39530878992884
// MI455X (gfx1250) — matrix operand order repaired and checked by reading, not compiled, not run on hardware
//
#include <hip/hip_runtime.h>
#include <hip/hip_bf16.h>

typedef __attribute__((ext_vector_type(16))) _Float16 v16h;
typedef __attribute__((ext_vector_type(8)))  float    v8f;

#define C_CH    128
#define HEADS   4
#define DHEAD   32
#define SEQ     4096   // 64*64
#define BATCH   4
#define GROUPS  8
#define QK_SCALE 0.17677669529663687f  // 32^-0.5

#if __has_builtin(__builtin_amdgcn_global_load_async_to_lds_b128)
#define HAS_ASYNC_LDS 1
#endif

union AF { v16h h; unsigned int u[8]; };
union BF { v16h h; unsigned int u[8]; };

// vector-of-4-int type matching the async-copy builtin's expected pointee
typedef __attribute__((vector_size(16))) int av4i;

__device__ __forceinline__ v8f wmma_f16(const v16h& a, const v16h& b, v8f c) {
    return __builtin_amdgcn_wmma_f32_16x16x32_f16(false, a, false, b, (short)0, c, false, false);
}

// A-fragment (16x32 f16, M x K): lane<16 -> row M=lane, K={0..7,16..23}; lane>=16 same row, K offset +8.
__device__ __forceinline__ int a_koff(int v, int lh) { return (v < 4 ? 2*v : 8 + 2*v) + 8*lh; }
// B-fragment (32x16 f16, K x N): lane%16 = N, K={0..7,16..23}; lane>=16 K offset +8 (same as A).
__device__ __forceinline__ int b_koff(int v, int lh) { return (v < 4 ? 2*v : 8 + 2*v) + 8*lh; }

__device__ __forceinline__ void async_cp16(const _Float16* g, _Float16* l) {
#ifdef HAS_ASYNC_LDS
    __builtin_amdgcn_global_load_async_to_lds_b128(
        (__attribute__((address_space(1))) av4i*)g,
        (__attribute__((address_space(3))) av4i*)l, 0, 0);
#else
    *(uint4*)l = *(const uint4*)g;
#endif
}

__device__ __forceinline__ void wait_async() {
#ifdef HAS_ASYNC_LDS
#if __has_builtin(__builtin_amdgcn_s_wait_asynccnt)
    __builtin_amdgcn_s_wait_asynccnt(0);
#else
    asm volatile("s_wait_asynccnt 0" ::: "memory");
#endif
#endif
}

// ---------------------------------------------------------------- group norm stats
__global__ void k_gnstats(const float* __restrict__ x, float* __restrict__ stats) {
    const int bg = blockIdx.x;                    // 0..31  (b*8+g)
    const float* p = x + (size_t)bg * 65536;      // 16 ch * 4096 contiguous
    float s = 0.f, ss = 0.f;
    for (int i = threadIdx.x; i < 65536; i += 256) { float v = p[i]; s += v; ss += v*v; }
    __shared__ float sh[512];
    sh[threadIdx.x] = s; sh[256 + threadIdx.x] = ss;
    __syncthreads();
    for (int st = 128; st > 0; st >>= 1) {
        if (threadIdx.x < st) {
            sh[threadIdx.x] += sh[threadIdx.x + st];
            sh[256 + threadIdx.x] += sh[256 + threadIdx.x + st];
        }
        __syncthreads();
    }
    if (threadIdx.x == 0) {
        float mean = sh[0] * (1.f / 65536.f);
        float var  = sh[256] * (1.f / 65536.f) - mean * mean;
        stats[bg] = mean;
        stats[32 + bg] = rsqrtf(var + 1e-5f);
    }
}

// ------------------------------------------------ normalize + transpose to f16 [b, token, c]
__global__ void k_gnnorm(const float* __restrict__ x, const float* __restrict__ stats,
                         const float* __restrict__ nw, const float* __restrict__ nb,
                         _Float16* __restrict__ xn) {
    int idx = blockIdx.x * 256 + threadIdx.x;     // 2,097,152 elements
    int b = idx >> 19, rem = idx & 524287;
    int c = rem >> 12, t = rem & 4095;
    int bg = b * GROUPS + (c >> 4);
    float v = (x[idx] - stats[bg]) * stats[32 + bg] * nw[c] + nb[c];
    xn[((size_t)b * SEQ + t) * C_CH + c] = (_Float16)v;
}

// ------------------------------------------------------------ fp32 -> f16 weight convert
__global__ void k_wconv(const float* __restrict__ qkvw, const float* __restrict__ projw,
                        _Float16* __restrict__ qkvh, _Float16* __restrict__ projh) {
    int i = blockIdx.x * 256 + threadIdx.x;       // 65536
    if (i < 49152) qkvh[i] = (_Float16)qkvw[i];
    else { int j = i - 49152; projh[j] = (_Float16)projw[j]; }
}

// ---------------------------------------------------- QKV GEMM: [384x128] x [128 x tokens]
__global__ void k_qkv(const _Float16* __restrict__ w, const float* __restrict__ bias,
                      const _Float16* __restrict__ xn,
                      _Float16* __restrict__ q, _Float16* __restrict__ k,
                      _Float16* __restrict__ vT) {
    const int lane = threadIdx.x & 31, wave = threadIdx.x >> 5;
    const int l16 = lane & 15, lh = lane >> 4;
    const int tile = blockIdx.x * 8 + wave;       // 24576 tiles
    const int o0 = (tile >> 10) * 16;             // out channel tile
    const int tok0 = (tile & 1023) * 16;          // global token tile
    const int b = tok0 >> 12, tl = tok0 & 4095;
    const _Float16* xb = xn + (size_t)b * SEQ * C_CH;
    v8f acc = {};
    #pragma unroll
    for (int kc = 0; kc < 4; ++kc) {
        AF a; BF bf;
        #pragma unroll
        for (int v = 0; v < 8; ++v) {
            a.u[v]  = *(const unsigned int*)(w  + (size_t)(o0 + l16) * C_CH + kc*32 + a_koff(v, lh));
            bf.u[v] = *(const unsigned int*)(xb + (size_t)(tl + l16) * C_CH + kc*32 + b_koff(v, lh));
        }
        acc = wmma_f16(a.h, bf.h, acc);
    }
    const int token = tl + l16;
    #pragma unroll
    for (int r = 0; r < 8; ++r) {
        int o = o0 + r + 8*lh;                    // out channel 0..383
        float val = acc[r] + bias[o];
        int mtx = o >> 7, head = (o >> 5) & 3, dd = o & 31;
        int bh = b * HEADS + head;
        if (mtx == 0)      q[((size_t)bh * SEQ + token) * DHEAD + dd] = (_Float16)(val * QK_SCALE);
        else if (mtx == 1) k[((size_t)bh * SEQ + token) * DHEAD + dd] = (_Float16)val;
        else               vT[((size_t)bh * DHEAD + dd) * SEQ + token] = (_Float16)val;
    }
}

// ---------------------------------------------------------------- flash attention
// 1 block = 4 waves on the SAME (b,head), 64 consecutive queries.
// K/V chunks (32 keys) staged once per block in LDS via async copy, double buffered.
__global__ void __launch_bounds__(128)
k_flash(const _Float16* __restrict__ qf, const _Float16* __restrict__ kf,
        const _Float16* __restrict__ vf, _Float16* __restrict__ att) {
    __shared__ _Float16 lds_k[2][32 * 32];   // [key][d]
    __shared__ _Float16 lds_v[2][32 * 32];   // [d][key]
    __shared__ _Float16 lds_p[4][16 * 32];

    const int tid = threadIdx.x;
    const int lane = tid & 31, wave = tid >> 5;
    const int l16 = lane & 15, lh = lane >> 4;
    const int bh = blockIdx.x >> 6;               // 16 batch-heads
    const int q0 = ((blockIdx.x & 63) << 6) + (wave << 4);
    const _Float16* qb = qf + (size_t)bh * SEQ * DHEAD;
    const _Float16* kb = kf + (size_t)bh * SEQ * DHEAD;
    const _Float16* vb = vf + (size_t)bh * DHEAD * SEQ;

    // cooperative stage of one 32-key chunk: 128 threads x 16B each for K and V
    const int srow = tid >> 2, sgrp = (tid & 3) << 3;   // row 0..31, half-offset 0/8/16/24

    AF qa;
    #pragma unroll
    for (int v = 0; v < 8; ++v)
        qa.u[v] = *(const unsigned int*)(qb + (size_t)(q0 + l16) * DHEAD + a_koff(v, lh));

    v8f o0 = {}, o1 = {};
    float m_r[8], l_r[8];
    #pragma unroll
    for (int r = 0; r < 8; ++r) { m_r[r] = -3.0e38f; l_r[r] = 0.f; }

    // prologue: stage chunk 0 into buffer 0
    async_cp16(kb + (size_t)srow * DHEAD + sgrp,      &lds_k[0][srow * 32 + sgrp]);
    async_cp16(vb + (size_t)srow * SEQ + sgrp,        &lds_v[0][srow * 32 + sgrp]);
    wait_async();
    __syncthreads();

    int buf = 0;
    for (int k0 = 0; k0 < SEQ; k0 += 32) {
        // prefetch next chunk into the other buffer (overlaps with compute below)
        if (k0 + 32 < SEQ) {
            async_cp16(kb + (size_t)(k0 + 32 + srow) * DHEAD + sgrp, &lds_k[buf ^ 1][srow * 32 + sgrp]);
            async_cp16(vb + (size_t)srow * SEQ + (k0 + 32) + sgrp,   &lds_v[buf ^ 1][srow * 32 + sgrp]);
        }

        const _Float16* lk = &lds_k[buf][0];
        const _Float16* lv = &lds_v[buf][0];

        BF kb0, kb1;
        #pragma unroll
        for (int v = 0; v < 8; ++v) {
            int ko = b_koff(v, lh);
            kb0.u[v] = *(const unsigned int*)(lk + l16 * 32 + ko);          // keys 0..15
            kb1.u[v] = *(const unsigned int*)(lk + (16 + l16) * 32 + ko);   // keys 16..31
        }
        v8f s0 = {}, s1 = {};
        s0 = wmma_f16(qa.h, kb0.h, s0);
        s1 = wmma_f16(qa.h, kb1.h, s1);

        // online softmax (row M = r + 8*lh, col N = l16)
        float alpha[8];
        #pragma unroll
        for (int r = 0; r < 8; ++r) {
            float t = fmaxf(s0[r], s1[r]);
            t = fmaxf(t, __shfl_xor(t, 1, 32));
            t = fmaxf(t, __shfl_xor(t, 2, 32));
            t = fmaxf(t, __shfl_xor(t, 4, 32));
            t = fmaxf(t, __shfl_xor(t, 8, 32));
            float mn = fmaxf(m_r[r], t);
            alpha[r] = __expf(m_r[r] - mn);
            m_r[r] = mn;
            float p0 = __expf(s0[r] - mn);
            float p1 = __expf(s1[r] - mn);
            s0[r] = p0; s1[r] = p1;
            float ps = p0 + p1;
            ps += __shfl_xor(ps, 1, 32);
            ps += __shfl_xor(ps, 2, 32);
            ps += __shfl_xor(ps, 4, 32);
            ps += __shfl_xor(ps, 8, 32);
            l_r[r] = l_r[r] * alpha[r] + ps;
            o0[r] *= alpha[r];
            o1[r] *= alpha[r];
        }

        // P (C-layout) -> LDS -> A-fragment
        _Float16* lp = &lds_p[wave][0];
        #pragma unroll
        for (int r = 0; r < 8; ++r) {
            int row = r + 8*lh;
            lp[row * 32 + l16]      = (_Float16)s0[r];
            lp[row * 32 + 16 + l16] = (_Float16)s1[r];
        }
        asm volatile("s_wait_dscnt 0" ::: "memory");
        AF pa;
        #pragma unroll
        for (int v = 0; v < 8; ++v)
            pa.u[v] = *(const unsigned int*)(lp + l16 * 32 + a_koff(v, lh));

        BF vb0, vb1;
        #pragma unroll
        for (int v = 0; v < 8; ++v) {
            int ko = b_koff(v, lh);
            vb0.u[v] = *(const unsigned int*)(lv + l16 * 32 + ko);          // dims 0..15
            vb1.u[v] = *(const unsigned int*)(lv + (16 + l16) * 32 + ko);   // dims 16..31
        }
        o0 = wmma_f16(pa.h, vb0.h, o0);
        o1 = wmma_f16(pa.h, vb1.h, o1);

        wait_async();          // own wave's async copies for next chunk done
        __syncthreads();       // all waves' copies visible, all done reading buf
        buf ^= 1;
    }

    const int b = bh >> 2, head = bh & 3;
    #pragma unroll
    for (int r = 0; r < 8; ++r) {
        float inv = 1.0f / l_r[r];
        int token = q0 + r + 8*lh;
        size_t base = ((size_t)b * SEQ + token) * C_CH + head * DHEAD;
        att[base + l16]      = (_Float16)(o0[r] * inv);
        att[base + 16 + l16] = (_Float16)(o1[r] * inv);
    }
}

// ---------------------------------- proj GEMM [128x128] x [128 x tokens] + bias + residual
__global__ void k_proj(const _Float16* __restrict__ w, const float* __restrict__ bias,
                       const _Float16* __restrict__ att, const float* __restrict__ x,
                       float* __restrict__ out) {
    const int lane = threadIdx.x & 31, wave = threadIdx.x >> 5;
    const int l16 = lane & 15, lh = lane >> 4;
    const int tile = blockIdx.x * 8 + wave;       // 8192 tiles
    const int o0 = (tile >> 10) * 16;
    const int tok0 = (tile & 1023) * 16;
    const int b = tok0 >> 12, tl = tok0 & 4095;
    const _Float16* ab = att + (size_t)b * SEQ * C_CH;
    v8f acc = {};
    #pragma unroll
    for (int kc = 0; kc < 4; ++kc) {
        AF a; BF bf;
        #pragma unroll
        for (int v = 0; v < 8; ++v) {
            a.u[v]  = *(const unsigned int*)(w  + (size_t)(o0 + l16) * C_CH + kc*32 + a_koff(v, lh));
            bf.u[v] = *(const unsigned int*)(ab + (size_t)(tl + l16) * C_CH + kc*32 + b_koff(v, lh));
        }
        acc = wmma_f16(a.h, bf.h, acc);
    }
    const int token = tl + l16;
    #pragma unroll
    for (int r = 0; r < 8; ++r) {
        int o = o0 + r + 8*lh;
        size_t xi = ((size_t)(b * C_CH + o)) * SEQ + token;
        out[xi] = acc[r] + bias[o] + x[xi];
    }
}

extern "C" void kernel_launch(void* const* d_in, const int* in_sizes, int n_in,
                              void* d_out, int out_size, void* d_ws, size_t ws_size,
                              hipStream_t stream) {
    const float* x      = (const float*)d_in[0];
    const float* norm_w = (const float*)d_in[1];
    const float* norm_b = (const float*)d_in[2];
    const float* qkv_w  = (const float*)d_in[3];
    const float* qkv_b  = (const float*)d_in[4];
    const float* proj_w = (const float*)d_in[5];
    const float* proj_b = (const float*)d_in[6];
    float* out = (float*)d_out;

    char* ws = (char*)d_ws;
    float*    stats  = (float*)(ws + 0);                       //   256 B
    _Float16* qkvw_h = (_Float16*)(ws + 256);                  //  96 KB
    _Float16* projw_h= (_Float16*)(ws + 98560);                //  32 KB
    _Float16* xn     = (_Float16*)(ws + 131328);               //   4 MB
    _Float16* qf     = (_Float16*)(ws + 4325632);              //   4 MB
    _Float16* kf     = (_Float16*)(ws + 8519936);              //   4 MB
    _Float16* vT     = (_Float16*)(ws + 12714240);             //   4 MB
    _Float16* att    = (_Float16*)(ws + 16908544);             //   4 MB

    k_gnstats<<<32, 256, 0, stream>>>(x, stats);
    k_wconv<<<256, 256, 0, stream>>>(qkv_w, proj_w, qkvw_h, projw_h);
    k_gnnorm<<<8192, 256, 0, stream>>>(x, stats, norm_w, norm_b, xn);
    k_qkv<<<3072, 256, 0, stream>>>(qkvw_h, qkv_b, xn, qf, kf, vT);
    k_flash<<<1024, 128, 0, stream>>>(qf, kf, vT, att);
    k_proj<<<1024, 256, 0, stream>>>(projw_h, proj_b, att, x, out);
}
